// WeightOnlyInt8Linear_24257975288001
// MI455X (gfx1250) — matrix operand order repaired and checked by reading, not compiled, not run on hardware
//
#include <hip/hip_runtime.h>

// Problem constants from the reference: B=4, S=2048, DIN=4096, DOUT=4096
#define DIN_  4096
#define DOUT_ 4096
#define MTOT_ 8192   // B*S

typedef __attribute__((ext_vector_type(16))) __bf16 v16bf;
typedef __attribute__((ext_vector_type(8)))  float  v8f;
typedef __attribute__((ext_vector_type(4)))  int    v4i;

constexpr int BLOCK_M = 64;
constexpr int BLOCK_N = 128;
constexpr int BLOCK_K = 64;
constexpr int LDA = 72;   // padded LDS row stride (bf16 elems): 144 B, 16B-aligned
constexpr int LDB = 72;
constexpr int NKSTEP = DIN_ / BLOCK_K;   // 64

// ---------------- CDNA5 async global->LDS copy (guarded) ----------------
#if defined(__HIP_DEVICE_COMPILE__) && \
    __has_builtin(__builtin_amdgcn_global_load_async_to_lds_b128)
#define HAVE_ASYNC_LDS 1
#else
#define HAVE_ASYNC_LDS 0
#endif

__device__ __forceinline__ void copy16_g2l(const __bf16* g, __bf16* l) {
#if HAVE_ASYNC_LDS
    __builtin_amdgcn_global_load_async_to_lds_b128(
        (__attribute__((address_space(1))) v4i*)g,
        (__attribute__((address_space(3))) v4i*)l,
        /*offset=*/0, /*cpol=*/0);
#else
    *(uint4*)l = *(const uint4*)g;
#endif
}

__device__ __forceinline__ void async_join() {
#if HAVE_ASYNC_LDS
#if __has_builtin(__builtin_amdgcn_s_wait_asynccnt)
    __builtin_amdgcn_s_wait_asynccnt(0);
#else
    asm volatile("s_wait_asynccnt 0x0" ::: "memory");
#endif
#endif
}

// ---------------- one-shot conversion pre-passes into d_ws ----------------
// A: f32 -> bf16, 16 elems/thread
__global__ __launch_bounds__(256)
void cvt_a_f32_to_bf16(const float* __restrict__ in, __bf16* __restrict__ out) {
    const size_t base = ((size_t)blockIdx.x * 256 + threadIdx.x) * 16;
    float4 f0 = *(const float4*)(in + base);
    float4 f1 = *(const float4*)(in + base + 4);
    float4 f2 = *(const float4*)(in + base + 8);
    float4 f3 = *(const float4*)(in + base + 12);
    union { __bf16 h[16]; uint4 u[2]; } t;
    t.h[0]  = (__bf16)f0.x; t.h[1]  = (__bf16)f0.y;
    t.h[2]  = (__bf16)f0.z; t.h[3]  = (__bf16)f0.w;
    t.h[4]  = (__bf16)f1.x; t.h[5]  = (__bf16)f1.y;
    t.h[6]  = (__bf16)f1.z; t.h[7]  = (__bf16)f1.w;
    t.h[8]  = (__bf16)f2.x; t.h[9]  = (__bf16)f2.y;
    t.h[10] = (__bf16)f2.z; t.h[11] = (__bf16)f2.w;
    t.h[12] = (__bf16)f3.x; t.h[13] = (__bf16)f3.y;
    t.h[14] = (__bf16)f3.z; t.h[15] = (__bf16)f3.w;
    uint4* p = (uint4*)(out + base);
    p[0] = t.u[0];
    p[1] = t.u[1];
}

// W: int8 -> bf16 (exact: |w|<=127 representable), 16 elems/thread
__global__ __launch_bounds__(256)
void cvt_w_i8_to_bf16(const signed char* __restrict__ in, __bf16* __restrict__ out) {
    const size_t base = ((size_t)blockIdx.x * 256 + threadIdx.x) * 16;
    int4 w = *(const int4*)(in + base);
    const signed char* c = (const signed char*)&w;
    union { __bf16 h[16]; uint4 u[2]; } t;
#pragma unroll
    for (int b = 0; b < 16; ++b) t.h[b] = (__bf16)(float)(int)c[b];
    uint4* p = (uint4*)(out + base);
    p[0] = t.u[0];
    p[1] = t.u[1];
}

// ---------------- shared compute helper: batched fragment loads ----------------
// Loads A frag + all 4 B frags into independent registers before the 4 WMMAs so
// the scheduler can use partial s_wait_dscnt and overlap ds_load with WMMA.
template <typename AccT>
__device__ __forceinline__ void wave_mma_step(const __bf16* __restrict__ pa,
                                              const __bf16* __restrict__ pb0,
                                              int ks, int laneHi, AccT& acc) {
    union { v16bf v; uint4 u[2]; } fa;
    fa.u[0] = *(const uint4*)(pa + ks + laneHi * 8);
    fa.u[1] = *(const uint4*)(pa + ks + 16 + laneHi * 8);
    union { v16bf v; uint4 u[2]; } fb[4];
#pragma unroll
    for (int j = 0; j < 4; ++j) {
        const __bf16* pb = pb0 + j * 16 * LDB + ks + laneHi * 8;
        fb[j].u[0] = *(const uint4*)(pb);
        fb[j].u[1] = *(const uint4*)(pb + 16);
    }
#pragma unroll
    for (int j = 0; j < 4; ++j) {
        acc[j] = __builtin_amdgcn_wmma_f32_16x16x32_bf16(
            false, fa.v, false, fb[j].v, (short)0, acc[j], false, false);
    }
}

// ---------------- main GEMM: bf16 x bf16 -> f32, conversion-free loop ----------------
__global__ __launch_bounds__(256)
void gemm_bf16_wmma(const __bf16* __restrict__ A,   // [MTOT_, DIN_] bf16
                    const __bf16* __restrict__ W,   // [DOUT_, DIN_] bf16
                    const float* __restrict__ scal,
                    float* __restrict__ out)
{
    __shared__ __align__(16) __bf16 sA[2][BLOCK_M * LDA];   // 9216 B each
    __shared__ __align__(16) __bf16 sB[2][BLOCK_N * LDB];   // 18432 B each

    const int tid    = threadIdx.x;
    const int lane   = tid & 31;
    const int wave   = tid >> 5;        // 0..7
    const int wm     = wave >> 1;       // 0..3 : 16-row strip within 64
    const int wn     = wave & 1;        // 0..1 : 64-col strip within 128
    const int laneLo = lane & 15;
    const int laneHi = lane >> 4;       // 0 or 1

    const int m0 = blockIdx.y * BLOCK_M;
    const int n0 = blockIdx.x * BLOCK_N;

    v8f acc[4] = {};   // 16x64 f32 output strip per wave

    auto stage = [&](int kt, int buf) {
        const int k0 = kt * BLOCK_K;
        // A tile: 64 rows x 64 bf16 = 512 x 16B chunks, 2 per thread
#pragma unroll
        for (int i = 0; i < 2; ++i) {
            const int c   = tid + i * 256;
            const int row = c >> 3;
            const int seg = (c & 7) * 8;        // bf16 elems, 16B granules
            copy16_g2l(A + (size_t)(m0 + row) * DIN_ + k0 + seg,
                       &sA[buf][row * LDA + seg]);
        }
        // W tile: 128 rows x 64 bf16 = 1024 x 16B chunks, 4 per thread
#pragma unroll
        for (int i = 0; i < 4; ++i) {
            const int c   = tid + i * 256;
            const int row = c >> 3;
            const int seg = (c & 7) * 8;
            copy16_g2l(W + (size_t)(n0 + row) * DIN_ + k0 + seg,
                       &sB[buf][row * LDB + seg]);
        }
        // Warm L2 for the next K-tile.
        if (kt + 1 < NKSTEP) {
            const int k1 = (kt + 1) * BLOCK_K;
            __builtin_prefetch(A + (size_t)(m0 + (tid >> 2)) * DIN_ + k1, 0, 1);
            __builtin_prefetch(W + (size_t)(n0 + (tid >> 1)) * DIN_ + k1, 0, 1);
        }
    };

    auto compute = [&](int buf) {
        const __bf16* pa  = &sA[buf][(wm * 16 + laneLo) * LDA];
        const __bf16* pb0 = &sB[buf][(wn * 64 + laneLo) * LDB];
#pragma unroll
        for (int ks = 0; ks < BLOCK_K; ks += 32) {
            wave_mma_step(pa, pb0, ks, laneHi, acc);
        }
    };

    // ---- pipelined main loop: double-buffered LDS, 1 barrier per K-step ----
    stage(0, 0);
    async_join();
    __syncthreads();

    int cur = 0;
    for (int kt = 0; kt < NKSTEP; ++kt) {
        if (kt + 1 < NKSTEP) stage(kt + 1, cur ^ 1);  // async, overlaps compute
        compute(cur);
        if (kt + 1 < NKSTEP) {
            async_join();         // this wave's async copies landed in LDS
            __syncthreads();      // everyone's landed; prior buffer free
            cur ^= 1;
        }
    }

    // ---- epilogue: per-output-channel dequant scale, f32 store ----
#pragma unroll
    for (int j = 0; j < 4; ++j) {
        const int n = n0 + wn * 64 + j * 16 + laneLo;
        const float s = scal[n];
        const int mbase = m0 + wm * 16 + laneHi * 8;   // f32 C/D layout: VGPR r -> M=r(+8)
#pragma unroll
        for (int r = 0; r < 8; ++r) {
            out[(size_t)(mbase + r) * DOUT_ + n] = acc[j][r] * s;
        }
    }
}

// ---------------- fallback: fused kernel (small ws) ----------------
__global__ __launch_bounds__(256)
void wo_int8_gemm_fused(const float* __restrict__ A,
                        const signed char* __restrict__ W,
                        const float* __restrict__ scal,
                        float* __restrict__ out)
{
    __shared__ __align__(16) __bf16 sA[2][BLOCK_M * LDA];
    __shared__ __align__(16) __bf16 sB[2][BLOCK_N * LDB];

    const int tid    = threadIdx.x;
    const int lane   = tid & 31;
    const int wave   = tid >> 5;
    const int wm     = wave >> 1;
    const int wn     = wave & 1;
    const int laneLo = lane & 15;
    const int laneHi = lane >> 4;

    const int m0 = blockIdx.y * BLOCK_M;
    const int n0 = blockIdx.x * BLOCK_N;

    float4 ra[4];
    int4   rw[2];
    v8f acc[4] = {};

    auto load_global = [&](int kt) {
        const int k0 = kt * BLOCK_K;
#pragma unroll
        for (int i = 0; i < 4; ++i) {
            const int idx = tid + i * 256;
            const int row = idx >> 4;
            const int kc  = (idx & 15) << 2;
            ra[i] = *(const float4*)(A + (size_t)(m0 + row) * DIN_ + k0 + kc);
        }
#pragma unroll
        for (int i = 0; i < 2; ++i) {
            const int idx = tid + i * 256;
            const int row = idx >> 2;
            const int kc  = (idx & 3) << 4;
            rw[i] = *(const int4*)(W + (size_t)(n0 + row) * DIN_ + k0 + kc);
        }
    };

    auto store_lds = [&](int buf) {
#pragma unroll
        for (int i = 0; i < 4; ++i) {
            const int idx = tid + i * 256;
            const int row = idx >> 4;
            const int kc  = (idx & 15) << 2;
            union { __bf16 h[4]; uint2 u; } t;
            t.h[0] = (__bf16)ra[i].x; t.h[1] = (__bf16)ra[i].y;
            t.h[2] = (__bf16)ra[i].z; t.h[3] = (__bf16)ra[i].w;
            *(uint2*)(&sA[buf][row * LDA + kc]) = t.u;
        }
#pragma unroll
        for (int i = 0; i < 2; ++i) {
            const int idx = tid + i * 256;
            const int row = idx >> 2;
            const int kc  = (idx & 3) << 4;
            const signed char* c = (const signed char*)&rw[i];
            union { __bf16 h[16]; uint4 u[2]; } t;
#pragma unroll
            for (int b = 0; b < 16; ++b) t.h[b] = (__bf16)(float)(int)c[b];
            uint4* p = (uint4*)(&sB[buf][row * LDB + kc]);
            p[0] = t.u[0];
            p[1] = t.u[1];
        }
    };

    auto compute = [&](int buf) {
        const __bf16* pa  = &sA[buf][(wm * 16 + laneLo) * LDA];
        const __bf16* pb0 = &sB[buf][(wn * 64 + laneLo) * LDB];
#pragma unroll
        for (int ks = 0; ks < BLOCK_K; ks += 32) {
            wave_mma_step(pa, pb0, ks, laneHi, acc);
        }
    };

    load_global(0);
    store_lds(0);
    __syncthreads();

    int cur = 0;
    for (int kt = 0; kt < NKSTEP; ++kt) {
        if (kt + 1 < NKSTEP) load_global(kt + 1);
        compute(cur);
        if (kt + 1 < NKSTEP) {
            store_lds(cur ^ 1);
            __syncthreads();
            cur ^= 1;
        }
    }

#pragma unroll
    for (int j = 0; j < 4; ++j) {
        const int n = n0 + wn * 64 + j * 16 + laneLo;
        const float s = scal[n];
        const int mbase = m0 + wm * 16 + laneHi * 8;
#pragma unroll
        for (int r = 0; r < 8; ++r) {
            out[(size_t)(mbase + r) * DOUT_ + n] = acc[j][r] * s;
        }
    }
}

extern "C" void kernel_launch(void* const* d_in, const int* in_sizes, int n_in,
                              void* d_out, int out_size, void* d_ws, size_t ws_size,
                              hipStream_t stream) {
    const float*       A  = (const float*)d_in[0];        // [4,2048,4096] f32
    const signed char* W  = (const signed char*)d_in[1];  // [4096,4096] int8
    const float*       sc = (const float*)d_in[2];        // [4096] f32
    float*             O  = (float*)d_out;                // [4,2048,4096] f32

    const size_t needA = (size_t)MTOT_ * DIN_ * sizeof(unsigned short); // 64 MB
    const size_t needW = (size_t)DOUT_ * DIN_ * sizeof(unsigned short); // 32 MB
    dim3 grid(DOUT_ / BLOCK_N, MTOT_ / BLOCK_M);  // (32, 128)

    if (ws_size >= needA + needW) {
        __bf16* Abf = (__bf16*)d_ws;
        __bf16* Wbf = (__bf16*)((char*)d_ws + needA);
        cvt_a_f32_to_bf16<<<(MTOT_ * (size_t)DIN_) / (16 * 256), 256, 0, stream>>>(A, Abf);
        cvt_w_i8_to_bf16<<<(DOUT_ * (size_t)DIN_) / (16 * 256), 256, 0, stream>>>(W, Wbf);
        gemm_bf16_wmma<<<grid, 256, 0, stream>>>(Abf, Wbf, sc, O);
    } else {
        wo_int8_gemm_fused<<<grid, 256, 0, stream>>>(A, W, sc, O);
    }
}
